// UpBlock_4638564680291
// MI455X (gfx1250) — matrix operand order repaired and checked by reading, not compiled, not run on hardware
//
#include <hip/hip_runtime.h>
#include <math.h>

typedef __bf16 bf16_t;
typedef __attribute__((ext_vector_type(16))) __bf16 v16bf;
typedef __attribute__((ext_vector_type(8)))  __bf16 v8bf;
typedef __attribute__((ext_vector_type(8)))  float  v8f;
typedef int v4i __attribute__((ext_vector_type(4)));

// Problem constants (match reference setup_inputs)
#define BB    16
#define NN    4096
#define MM    1024
#define DORI  256
#define DSUB  512
#define DIN   768      // DORI + DSUB
#define DOUT  512

#if __has_builtin(__builtin_amdgcn_global_load_async_to_lds_b128)
#define HAVE_ASYNC_LDS 1
#else
#define HAVE_ASYNC_LDS 0
#endif

__device__ __forceinline__ void copy16_g2l(bf16_t* lds, const bf16_t* g) {
#if HAVE_ASYNC_LDS
    __builtin_amdgcn_global_load_async_to_lds_b128(
        (__attribute__((address_space(1))) v4i*)g,
        (__attribute__((address_space(3))) v4i*)lds, 0, 0);
#else
    *(v8bf*)lds = *(const v8bf*)g;
#endif
}

__device__ __forceinline__ void wait_async_copies() {
#if HAVE_ASYNC_LDS
#if __has_builtin(__builtin_amdgcn_s_wait_asynccnt)
    __builtin_amdgcn_s_wait_asynccnt(0);
#else
    asm volatile("s_wait_asynccnt 0x0" ::: "memory");
#endif
#endif
}

// ---------------------------------------------------------------- utilities
__global__ __launch_bounds__(256) void cvt_f32_to_bf16(const float* __restrict__ s,
                                                       bf16_t* __restrict__ d, int n) {
    int i = blockIdx.x * 256 + threadIdx.x;
    if (i < n) d[i] = (bf16_t)s[i];
}

// Transpose f32 [D][L] (per batch) -> bf16 dst[l][ldd] at column offset co.
// Requires D,L multiples of 32.  block = 256 (32x8), grid = (L/32, D/32, B)
__global__ __launch_bounds__(256) void transpose_f32_bf16(const float* __restrict__ src,
                                                          bf16_t* __restrict__ dst,
                                                          int D, int L, int ldd, int co) {
    __shared__ float tile[32][33];
    int b = blockIdx.z;
    src += (size_t)b * D * L;
    dst += (size_t)b * L * ldd;
    int l0 = blockIdx.x * 32, d0 = blockIdx.y * 32;
    int tx = threadIdx.x & 31, ty = threadIdx.x >> 5;
#pragma unroll
    for (int i = 0; i < 32; i += 8)
        tile[ty + i][tx] = src[(size_t)(d0 + ty + i) * L + l0 + tx];
    __syncthreads();
#pragma unroll
    for (int i = 0; i < 32; i += 8)
        dst[(size_t)(l0 + ty + i) * ldd + co + d0 + tx] = (bf16_t)tile[tx][ty + i];
}

// ---------------------------------------------------------------- KNN + interpolation
__global__ __launch_bounds__(256) void knn_interp(const float* __restrict__ ori_xyz,  // [B][3][N]
                                                  const float* __restrict__ sub_xyz,  // [B][3][M]
                                                  const bf16_t* __restrict__ subxT,   // [B][M][DSUB]
                                                  bf16_t* __restrict__ Xcat) {        // [B][N][DIN]
    int b    = blockIdx.y;
    int wave = threadIdx.x >> 5;
    int lane = threadIdx.x & 31;
    int n    = blockIdx.x * 8 + wave;

    const float* oz = ori_xyz + (size_t)b * 3 * NN;
    float qx = oz[n], qy = oz[NN + n], qz = oz[2 * NN + n];

    const float* sz = sub_xyz + (size_t)b * 3 * MM;
    float d0 = 1e30f, d1 = 1e30f, d2 = 1e30f;
    int   i0 = 0, i1 = 0, i2 = 0;

#define INS3(dd, ii)                                                   \
    do {                                                               \
        float _d = (dd); int _i = (ii);                                \
        if (_d < d2) {                                                 \
            if (_d < d0)      { d2=d1;i2=i1; d1=d0;i1=i0; d0=_d;i0=_i; } \
            else if (_d < d1) { d2=d1;i2=i1; d1=_d;i1=_i; }            \
            else              { d2=_d;i2=_i; }                         \
        }                                                              \
    } while (0)

    for (int s = lane; s < MM; s += 32) {
        float dx = qx - sz[s];
        float dy = qy - sz[MM + s];
        float dz = qz - sz[2 * MM + s];
        INS3(dx * dx + dy * dy + dz * dz, s);
    }
#pragma unroll
    for (int off = 16; off; off >>= 1) {
        float e0 = __shfl_xor(d0, off, 32);
        float e1 = __shfl_xor(d1, off, 32);
        float e2 = __shfl_xor(d2, off, 32);
        int   j0 = __shfl_xor(i0, off, 32);
        int   j1 = __shfl_xor(i1, off, 32);
        int   j2 = __shfl_xor(i2, off, 32);
        INS3(e0, j0); INS3(e1, j1); INS3(e2, j2);
    }
#undef INS3

    float w0 = 1.0f / (sqrtf(fmaxf(d0, 1e-12f)) + 1e-5f);
    float w1 = 1.0f / (sqrtf(fmaxf(d1, 1e-12f)) + 1e-5f);
    float w2 = 1.0f / (sqrtf(fmaxf(d2, 1e-12f)) + 1e-5f);
    float ws = 1.0f / (w0 + w1 + w2);
    w0 *= ws; w1 *= ws; w2 *= ws;

    const bf16_t* s0 = subxT + ((size_t)b * MM + i0) * DSUB;
    const bf16_t* s1 = subxT + ((size_t)b * MM + i1) * DSUB;
    const bf16_t* s2 = subxT + ((size_t)b * MM + i2) * DSUB;
    bf16_t* xc = Xcat + ((size_t)b * NN + n) * DIN + DORI;
#pragma unroll 4
    for (int c = lane; c < DSUB; c += 32)
        xc[c] = (bf16_t)(w0 * (float)s0[c] + w1 * (float)s1[c] + w2 * (float)s2[c]);
}

// ---------------------------------------------------------------- bf16 WMMA GEMM
// Y[b][n][m] = sum_k W[m][k] * X[b][n][k]
// block = 128 (4 waves).  Block tile 64(M) x 128(N); wave w covers 32 columns.
// W slice (64 x Kd) staged in LDS as double-buffered 64x64 K-panels via async
// global->LDS copies; X fragments stream from global with 1-step reg prefetch.
#define KP      64
#define LDSK    (KP + 8)   // padded row: conflict-free ds_load_b128 frag reads
__global__ __launch_bounds__(128) void gemm_bf16(const bf16_t* __restrict__ W,  // [O][Kd]
                                                 const bf16_t* __restrict__ X,  // [B][Npts][Kd]
                                                 bf16_t* __restrict__ Y,        // [B][Npts][O]
                                                 int O, int Kd, int Npts) {
    __shared__ __align__(16) bf16_t As[2][64][LDSK];

    int b    = blockIdx.z;
    int n0   = blockIdx.x * 128;
    int m0   = blockIdx.y * 64;
    int wave = threadIdx.x >> 5;
    int lane = threadIdx.x & 31;
    int nw   = n0 + wave * 32;

    int lcol  = lane & 15;
    int khalf = (lane >> 4) * 16;  // B-frag: lanes 16-31 hold K=16..31
    int kq    = (lane >> 4) * 16;  // A-frag: lanes 16-31 hold K=16..31
    int arow  = lane & 15;

    const bf16_t* xp0 = X + ((size_t)b * Npts + (nw + lcol)) * Kd + khalf;
    const bf16_t* xp1 = xp0 + (size_t)16 * Kd;

    const bf16_t* Wm = W + (size_t)m0 * Kd;
    int npanels = Kd / KP;

    // cooperative K-panel load: 64 rows x 64 cols bf16 = 512 16B chunks
    auto load_panel = [&](int p, int buf) {
        const bf16_t* src = Wm + p * KP;
        for (int idx = threadIdx.x; idx < 64 * (KP / 8); idx += 128) {
            int r = idx >> 3;
            int c = (idx & 7) * 8;
            copy16_g2l(&As[buf][r][c], src + (size_t)r * Kd + c);
        }
    };

    load_panel(0, 0);

    v8f acc[8] = {};
    v16bf bc0 = *(const v16bf*)(xp0);
    v16bf bc1 = *(const v16bf*)(xp1);

    wait_async_copies();
    __syncthreads();

    for (int p = 0; p < npanels; ++p) {
        if (p + 1 < npanels) load_panel(p + 1, (p + 1) & 1);
#pragma unroll
        for (int kk = 0; kk < 2; ++kk) {
            int k0 = p * KP + kk * 32;
            int kn = k0 + 32;
            if (kn >= Kd) kn = 0;  // harmless wrap on last step
            v16bf bn0 = *(const v16bf*)(xp0 + kn);
            v16bf bn1 = *(const v16bf*)(xp1 + kn);
#pragma unroll
            for (int f = 0; f < 4; ++f) {
                const bf16_t* ap = &As[p & 1][f * 16 + arow][kk * 32 + kq];
                union { v16bf v; struct { v8bf lo; v8bf hi; } s; } ua;
                ua.s.lo = *(const v8bf*)ap;
                ua.s.hi = *(const v8bf*)(ap + 8);
                acc[f]     = __builtin_amdgcn_wmma_f32_16x16x32_bf16(
                    false, ua.v, false, bc0, (short)0, acc[f], false, false);
                acc[4 + f] = __builtin_amdgcn_wmma_f32_16x16x32_bf16(
                    false, ua.v, false, bc1, (short)0, acc[4 + f], false, false);
            }
            bc0 = bn0; bc1 = bn1;
        }
        wait_async_copies();
        __syncthreads();
    }

    // C layout: VGPR r -> row m0+f*16+r (+8 for lanes>=16), col = group base + (lane&15).
    // Per lane the 8 rows are consecutive channels -> one b128 store each.
#pragma unroll
    for (int g = 0; g < 2; ++g) {
        bf16_t* yb = Y + ((size_t)b * Npts + (nw + g * 16 + lcol)) * O + m0 + (lane >> 4) * 8;
#pragma unroll
        for (int f = 0; f < 4; ++f) {
            v8bf out;
#pragma unroll
            for (int r = 0; r < 8; ++r) out[r] = (bf16_t)acc[4 * g + f][r];
            *(v8bf*)(yb + f * 16) = out;
        }
    }
}

// ---------------------------------------------------------------- BatchNorm pieces
__global__ __launch_bounds__(256) void bn_stats(const bf16_t* __restrict__ Y,
                                                float* __restrict__ stats, int rows) {
    int r0 = blockIdx.x * 32;
    int t  = threadIdx.x;
    float s0 = 0.f, q0 = 0.f, s1 = 0.f, q1 = 0.f;
    for (int r = 0; r < 32; ++r) {
        const bf16_t* row = Y + (size_t)(r0 + r) * DOUT;
        float a = (float)row[t];        s0 += a; q0 += a * a;
        float c = (float)row[t + 256];  s1 += c; q1 += c * c;
    }
    atomicAdd(&stats[t],              s0);
    atomicAdd(&stats[DOUT + t],       q0);
    atomicAdd(&stats[t + 256],        s1);
    atomicAdd(&stats[DOUT + t + 256], q1);
}

__global__ void bn_finalize(float* stats, float inv_count) {
    int t = threadIdx.x;  // 512 threads
    float mean = stats[t] * inv_count;
    float var  = stats[DOUT + t] * inv_count - mean * mean;
    stats[t]        = mean;
    stats[DOUT + t] = rsqrtf(var + 1e-5f);
}

__global__ __launch_bounds__(256) void bn_gelu_bf16(bf16_t* Y, const float* __restrict__ stats,
                                                    const float* __restrict__ gamma,
                                                    const float* __restrict__ beta) {
    size_t i = (size_t)blockIdx.x * 256 + threadIdx.x;
    int    c = (int)(i & (DOUT - 1));
    float  y = (float)Y[i];
    y = (y - stats[c]) * stats[DOUT + c] * gamma[c] + beta[c];
    Y[i] = (bf16_t)(0.5f * y * (1.0f + erff(y * 0.70710678118f)));
}

__global__ __launch_bounds__(256) void bn_gelu_out(const bf16_t* __restrict__ Y,  // [B][N][512]
                                                   const float* __restrict__ stats,
                                                   const float* __restrict__ gamma,
                                                   const float* __restrict__ beta,
                                                   float* __restrict__ out) {
    __shared__ float tile[32][33];
    int b  = blockIdx.z;
    int m0 = blockIdx.x * 32, n0 = blockIdx.y * 32;
    int tx = threadIdx.x & 31, ty = threadIdx.x >> 5;
    const bf16_t* yb = Y + (size_t)b * NN * DOUT;
#pragma unroll
    for (int i = 0; i < 32; i += 8) {
        int   m = m0 + tx;
        float y = (float)yb[(size_t)(n0 + ty + i) * DOUT + m];
        y = (y - stats[m]) * stats[DOUT + m] * gamma[m] + beta[m];
        tile[ty + i][tx] = 0.5f * y * (1.0f + erff(y * 0.70710678118f));
    }
    __syncthreads();
    float* ob = out + (size_t)b * DOUT * NN;
#pragma unroll
    for (int i = 0; i < 32; i += 8)
        ob[(size_t)(m0 + ty + i) * NN + n0 + tx] = tile[tx][ty + i];
}

// ---------------------------------------------------------------- launch
extern "C" void kernel_launch(void* const* d_in, const int* in_sizes, int n_in,
                              void* d_out, int out_size, void* d_ws, size_t ws_size,
                              hipStream_t stream) {
    const float* ori_x   = (const float*)d_in[0];
    const float* ori_xyz = (const float*)d_in[1];
    const float* sub_x   = (const float*)d_in[2];
    const float* sub_xyz = (const float*)d_in[3];
    const float* W1      = (const float*)d_in[4];
    const float* gamma1  = (const float*)d_in[5];
    const float* beta1   = (const float*)d_in[6];
    const float* W2      = (const float*)d_in[7];
    const float* gamma2  = (const float*)d_in[8];
    const float* beta2   = (const float*)d_in[9];
    float* out = (float*)d_out;

    char* wsb = (char*)d_ws;
    const size_t OFF_XCAT  = 0;                                      // [B][N][768]  bf16
    const size_t OFF_Y1    = OFF_XCAT + (size_t)BB * NN * DIN * 2;   // [B][N][512]  bf16
    const size_t OFF_SUBXT = OFF_Y1   + (size_t)BB * NN * DOUT * 2;  // [B][M][512]  bf16
    const size_t OFF_W1    = OFF_SUBXT + (size_t)BB * MM * DSUB * 2;
    const size_t OFF_W2    = OFF_W1 + (size_t)DOUT * DIN * 2;
    const size_t OFF_ST1   = OFF_W2 + (size_t)DOUT * DOUT * 2;
    const size_t OFF_ST2   = OFF_ST1 + 4096;
    const size_t OFF_Y2    = OFF_XCAT;  // Xcat dead after GEMM1 -> reuse

    bf16_t* Xcat  = (bf16_t*)(wsb + OFF_XCAT);
    bf16_t* Y1    = (bf16_t*)(wsb + OFF_Y1);
    bf16_t* Y2    = (bf16_t*)(wsb + OFF_Y2);
    bf16_t* subxT = (bf16_t*)(wsb + OFF_SUBXT);
    bf16_t* W1b   = (bf16_t*)(wsb + OFF_W1);
    bf16_t* W2b   = (bf16_t*)(wsb + OFF_W2);
    float*  st1   = (float*)(wsb + OFF_ST1);
    float*  st2   = (float*)(wsb + OFF_ST2);

    (void)hipMemsetAsync(wsb + OFF_ST1, 0, 8192, stream);

    cvt_f32_to_bf16<<<(DOUT * DIN + 255) / 256, 256, 0, stream>>>(W1, W1b, DOUT * DIN);
    cvt_f32_to_bf16<<<(DOUT * DOUT + 255) / 256, 256, 0, stream>>>(W2, W2b, DOUT * DOUT);

    transpose_f32_bf16<<<dim3(NN / 32, DORI / 32, BB), 256, 0, stream>>>(ori_x, Xcat, DORI, NN, DIN, 0);
    transpose_f32_bf16<<<dim3(MM / 32, DSUB / 32, BB), 256, 0, stream>>>(sub_x, subxT, DSUB, MM, DSUB, 0);

    knn_interp<<<dim3(NN / 8, BB), 256, 0, stream>>>(ori_xyz, sub_xyz, subxT, Xcat);

    gemm_bf16<<<dim3(NN / 128, DOUT / 64, BB), 128, 0, stream>>>(W1b, Xcat, Y1, DOUT, DIN, NN);

    bn_stats<<<(BB * NN) / 32, 256, 0, stream>>>(Y1, st1, BB * NN);
    bn_finalize<<<1, DOUT, 0, stream>>>(st1, 1.0f / (float)(BB * NN));
    bn_gelu_bf16<<<(unsigned)((size_t)BB * NN * DOUT / 256), 256, 0, stream>>>(Y1, st1, gamma1, beta1);

    gemm_bf16<<<dim3(NN / 128, DOUT / 64, BB), 128, 0, stream>>>(W2b, Y1, Y2, DOUT, DOUT, NN);

    bn_stats<<<(BB * NN) / 32, 256, 0, stream>>>(Y2, st2, BB * NN);
    bn_finalize<<<1, DOUT, 0, stream>>>(st2, 1.0f / (float)(BB * NN));
    bn_gelu_out<<<dim3(DOUT / 32, NN / 32, BB), 256, 0, stream>>>(Y2, st2, gamma2, beta2, out);
}
